// LinearCrossEntropyLoss_46651934769795
// MI455X (gfx1250) — matrix operand order repaired and checked by reading, not compiled, not run on hardware
//
#include <hip/hip_runtime.h>
#include <hip/hip_bf16.h>

typedef __attribute__((ext_vector_type(8)))  float  v8f;
typedef __attribute__((ext_vector_type(8)))  __bf16 v8bf;
typedef __attribute__((ext_vector_type(16))) __bf16 v16bf;

union BF16Frag { v16bf v; v8bf h[2]; };

#define MT        128    // rows per workgroup tile
#define NCOLS     256    // vocab columns per chunk
#define KSTEP     32     // bf16 WMMA K depth
#define ASTR      40     // LDS row stride in ushorts (16B aligned, conflict-free)
#define BSTR      40

__device__ __forceinline__ unsigned short f2bf(float f) {
  union { float f; unsigned int u; } x; x.f = f;
  unsigned int r = x.u + 0x7FFFu + ((x.u >> 16) & 1u);  // round-to-nearest-even
  return (unsigned short)(r >> 16);
}

// ---------------------------------------------------------------------------
// Kernel 1: fused GEMM (bf16 WMMA, double-buffered LDS) + online softmax.
// grid = (N/MT, nsplit), block = 256 (8 wave32s as 4(M=32) x 2(N=128)).
// Each wave: 2(M) x 8(N) = 16 WMMA tiles, 128 f32 acc VGPRs.
// ---------------------------------------------------------------------------
__global__ __launch_bounds__(256) void lce_gemm_stats(
    const float* __restrict__ H, const float* __restrict__ W,
    float* __restrict__ part_max, float* __restrict__ part_sum,
    int N, int D, int chunks_per_split)
{
  __shared__ unsigned short As[2][MT * ASTR];     // 2 x 10.0 KB
  __shared__ unsigned short Bs[2][NCOLS * BSTR];  // 2 x 20.0 KB
  __shared__ float wstat[8][32];
  __shared__ float chunkmax[MT];
  __shared__ float runm[MT];
  __shared__ float runs[MT];

  const int t    = threadIdx.x;
  const int lane = t & 31;
  const int wave = t >> 5;
  const int wm   = wave >> 1;          // 0..3 : row group of 32
  const int wn   = wave & 1;           // 0..1 : col group of 128
  const int half = lane >> 4;
  const int l16  = lane & 15;
  const int rowbase = blockIdx.x * MT;

  const int ldr = t >> 3;              // 0..31 : staging row
  const int ldk = (t & 7) << 2;        // 0,4,...,28 : staging k (float4)

  if (t < MT) { runm[t] = -__builtin_inff(); runs[t] = 0.0f; }

  const int nk = D / KSTEP;

  for (int cc = 0; cc < chunks_per_split; ++cc) {
    const int colbase = (blockIdx.y * chunks_per_split + cc) * NCOLS;

    v8f acc[16];
    #pragma unroll
    for (int j = 0; j < 16; ++j)
      #pragma unroll
      for (int e = 0; e < 8; ++e) acc[j][e] = 0.0f;

    float4 ra[4];
    float4 rb[8];
    auto gload = [&](int kt) {
      #pragma unroll
      for (int q = 0; q < 4; ++q)
        ra[q] = *(const float4*)&H[(size_t)(rowbase + ldr + 32 * q) * D + kt + ldk];
      #pragma unroll
      for (int i = 0; i < 8; ++i)
        rb[i] = *(const float4*)&W[(size_t)(colbase + i * 32 + ldr) * D + kt + ldk];
    };
    auto sstore = [&](int buf) {
      #pragma unroll
      for (int q = 0; q < 4; ++q) {
        unsigned short* d = &As[buf][(ldr + 32 * q) * ASTR + ldk];
        d[0] = f2bf(ra[q].x); d[1] = f2bf(ra[q].y);
        d[2] = f2bf(ra[q].z); d[3] = f2bf(ra[q].w);
      }
      #pragma unroll
      for (int i = 0; i < 8; ++i) {
        unsigned short* d = &Bs[buf][(i * 32 + ldr) * BSTR + ldk];
        d[0] = f2bf(rb[i].x); d[1] = f2bf(rb[i].y);
        d[2] = f2bf(rb[i].z); d[3] = f2bf(rb[i].w);
      }
    };

    // prologue: stage k-step 0 into buffer 0
    gload(0);
    sstore(0);

    for (int ki = 0; ki < nk; ++ki) {
      __syncthreads();                       // buf ready; prior reads done
      const int buf = ki & 1;
      if (ki + 1 < nk) gload((ki + 1) * KSTEP);   // overlap with WMMAs below

      BF16Frag a0, a1;
      const unsigned short* ap0 = &As[buf][(wm * 32 + l16) * ASTR + half * 8];
      const unsigned short* ap1 = ap0 + 16 * ASTR;
      a0.h[0] = *(const v8bf*)(ap0); a0.h[1] = *(const v8bf*)(ap0 + 16);
      a1.h[0] = *(const v8bf*)(ap1); a1.h[1] = *(const v8bf*)(ap1 + 16);

      BF16Frag bb[2];
      {
        const unsigned short* bp = &Bs[buf][(wn * 128 + l16) * BSTR + half * 8];
        bb[0].h[0] = *(const v8bf*)(bp); bb[0].h[1] = *(const v8bf*)(bp + 16);
      }
      #pragma unroll
      for (int j = 0; j < 8; ++j) {
        if (j < 7) {                         // prefetch next B fragment
          const unsigned short* bp =
              &Bs[buf][(wn * 128 + (j + 1) * 16 + l16) * BSTR + half * 8];
          bb[(j + 1) & 1].h[0] = *(const v8bf*)(bp);
          bb[(j + 1) & 1].h[1] = *(const v8bf*)(bp + 16);
        }
        acc[j] = __builtin_amdgcn_wmma_f32_16x16x32_bf16(
            false, a0.v, false, bb[j & 1].v, (short)0, acc[j], false, false);
        acc[8 + j] = __builtin_amdgcn_wmma_f32_16x16x32_bf16(
            false, a1.v, false, bb[j & 1].v, (short)0, acc[8 + j], false, false);
      }
      if (ki + 1 < nk) sstore(buf ^ 1);      // waits loadcnt here, after WMMAs
    }
    __syncthreads();

    // ---- per-row max over this 128x256 chunk (registers + wave32 shuffles) ----
    float rm[16];   // [im*8 + r]
    #pragma unroll
    for (int im = 0; im < 2; ++im)
      #pragma unroll
      for (int r = 0; r < 8; ++r) {
        float m = acc[im * 8][r];
        #pragma unroll
        for (int j = 1; j < 8; ++j) m = fmaxf(m, acc[im * 8 + j][r]);
        rm[im * 8 + r] = m;
      }
    #pragma unroll
    for (int off = 1; off < 16; off <<= 1)
      #pragma unroll
      for (int q = 0; q < 16; ++q)
        rm[q] = fmaxf(rm[q], __shfl_xor(rm[q], off, 32));
    if (l16 == 0) {
      #pragma unroll
      for (int im = 0; im < 2; ++im)
        #pragma unroll
        for (int r = 0; r < 8; ++r)
          wstat[wave][im * 16 + half * 8 + r] = rm[im * 8 + r];
    }
    __syncthreads();
    if (t < MT) {
      const int g = (t >> 5) * 2;   // the two waves covering this row group
      chunkmax[t] = fmaxf(wstat[g][t & 31], wstat[g + 1][t & 31]);
    }
    __syncthreads();
    // ---- per-row sum(exp(x - max)) over this chunk ----
    float rs[16];
    #pragma unroll
    for (int im = 0; im < 2; ++im)
      #pragma unroll
      for (int r = 0; r < 8; ++r) {
        const float m = chunkmax[wm * 32 + im * 16 + half * 8 + r];
        float s = 0.0f;
        #pragma unroll
        for (int j = 0; j < 8; ++j) s += __expf(acc[im * 8 + j][r] - m);
        rs[im * 8 + r] = s;
      }
    #pragma unroll
    for (int off = 1; off < 16; off <<= 1)
      #pragma unroll
      for (int q = 0; q < 16; ++q)
        rs[q] += __shfl_xor(rs[q], off, 32);
    if (l16 == 0) {
      #pragma unroll
      for (int im = 0; im < 2; ++im)
        #pragma unroll
        for (int r = 0; r < 8; ++r)
          wstat[wave][im * 16 + half * 8 + r] = rs[im * 8 + r];
    }
    __syncthreads();
    if (t < MT) {
      const int g  = (t >> 5) * 2;
      const float sc = wstat[g][t & 31] + wstat[g + 1][t & 31];
      const float mc = chunkmax[t];
      const float om = runm[t];
      const float nm = fmaxf(om, mc);
      runs[t] = runs[t] * __expf(om - nm) + sc * __expf(mc - nm);
      runm[t] = nm;
    }
    __syncthreads();
  }
  if (t < MT) {
    part_max[(size_t)blockIdx.y * N + rowbase + t] = runm[t];
    part_sum[(size_t)blockIdx.y * N + rowbase + t] = runs[t];
  }
}

// ---------------------------------------------------------------------------
// Kernel 2: merge split partials -> logsumexp per row.
// ---------------------------------------------------------------------------
__global__ __launch_bounds__(256) void lce_combine(
    const float* __restrict__ part_max, const float* __restrict__ part_sum,
    float* __restrict__ lse, int N, int nsplit)
{
  const int n = blockIdx.x * 256 + threadIdx.x;
  if (n >= N) return;
  float m = -__builtin_inff(), s = 0.0f;
  for (int sp = 0; sp < nsplit; ++sp) {
    const float pm = part_max[(size_t)sp * N + n];
    const float ps = part_sum[(size_t)sp * N + n];
    const float nm = fmaxf(m, pm);
    s = s * __expf(m - nm) + ps * __expf(pm - nm);
    m = nm;
  }
  lse[n] = m + logf(s);
}

// ---------------------------------------------------------------------------
// Kernel 3: one wave32 per row -> target logit (fp32 dot) and nll.
// ---------------------------------------------------------------------------
__global__ __launch_bounds__(256) void lce_target(
    const float* __restrict__ H, const float* __restrict__ W,
    const int* __restrict__ tgt, const float* __restrict__ lse,
    float* __restrict__ nll, float* __restrict__ cnt, int N, int D)
{
  const int lane = threadIdx.x & 31;
  const int n = (blockIdx.x * 256 + threadIdx.x) >> 5;
  if (n >= N) return;
  const int tg = tgt[n];
  if (tg == -100) {   // IGNORE_INDEX
    if (lane == 0) { nll[n] = 0.0f; cnt[n] = 0.0f; }
    return;
  }
  const float* hp = &H[(size_t)n * D];
  const float* wp = &W[(size_t)tg * D];
  float acc = 0.0f;
  for (int k = lane * 4; k < D; k += 128) {
    const float4 a = *(const float4*)&hp[k];
    const float4 b = *(const float4*)&wp[k];
    acc = fmaf(a.x, b.x, acc); acc = fmaf(a.y, b.y, acc);
    acc = fmaf(a.z, b.z, acc); acc = fmaf(a.w, b.w, acc);
  }
  #pragma unroll
  for (int off = 16; off > 0; off >>= 1) acc += __shfl_xor(acc, off, 32);
  if (lane == 0) { nll[n] = lse[n] - acc; cnt[n] = 1.0f; }
}

// ---------------------------------------------------------------------------
// Kernel 4: deterministic fixed-order reduction -> masked mean.
// ---------------------------------------------------------------------------
__global__ __launch_bounds__(256) void lce_reduce(
    const float* __restrict__ nll, const float* __restrict__ cnt,
    float* __restrict__ out, int N)
{
  __shared__ float ss[256];
  __shared__ float sc[256];
  float s = 0.0f, c = 0.0f;
  for (int i = threadIdx.x; i < N; i += 256) { s += nll[i]; c += cnt[i]; }
  ss[threadIdx.x] = s; sc[threadIdx.x] = c;
  __syncthreads();
  for (int st = 128; st > 0; st >>= 1) {
    if (threadIdx.x < (unsigned)st) {
      ss[threadIdx.x] += ss[threadIdx.x + st];
      sc[threadIdx.x] += sc[threadIdx.x + st];
    }
    __syncthreads();
  }
  if (threadIdx.x == 0) out[0] = (sc[0] > 0.0f) ? (ss[0] / sc[0]) : ss[0];
}

// ---------------------------------------------------------------------------
extern "C" void kernel_launch(void* const* d_in, const int* in_sizes, int n_in,
                              void* d_out, int out_size, void* d_ws, size_t ws_size,
                              hipStream_t stream) {
  const float* H = (const float*)d_in[0];   // [N, D] fp32
  const float* W = (const float*)d_in[1];   // [V, D] fp32
  const int*   T = (const int*)d_in[2];     // [N]
  float* out = (float*)d_out;

  const int N = in_sizes[2];                // B*S = 8192
  const int D = in_sizes[0] / N;            // 2048
  const int V = in_sizes[1] / D;            // 32000

  const int nchunks = V / NCOLS;            // 125
  int nsplit = nchunks < 25 ? nchunks : 25;
  while (nchunks % nsplit) --nsplit;        // 25 for V=32000
  const int cps = nchunks / nsplit;         // 5

  float* ws       = (float*)d_ws;
  float* part_max = ws;
  float* part_sum = part_max + (size_t)nsplit * N;
  float* lse      = part_sum + (size_t)nsplit * N;
  float* nll      = lse + N;
  float* cnt      = nll + N;

  dim3 g1(N / MT, nsplit);
  lce_gemm_stats<<<g1, 256, 0, stream>>>(H, W, part_max, part_sum, N, D, cps);
  lce_combine<<<(N + 255) / 256, 256, 0, stream>>>(part_max, part_sum, lse, N, nsplit);
  lce_target<<<(N * 32 + 255) / 256, 256, 0, stream>>>(H, W, T, lse, nll, cnt, N, D);
  lce_reduce<<<1, 256, 0, stream>>>(nll, cnt, out, N);
}
